// TiledSoftmax_68968584839721
// MI455X (gfx1250) — compile-verified, matrix operand order checked by reading, not run on hardware
//
#include <hip/hip_runtime.h>
#include <stdint.h>

// ---------------------------------------------------------------------------
// TiledSoftmax (LM head + row softmax) for gfx1250 / MI455X
//   logits = input[2048,2048] @ proj_weight[32000,2048]^T   (bf16 WMMA, f32 acc)
//   out    = softmax(logits, axis=-1)                       (in place on d_out)
// Pipeline: cvt W->bf16 (131MB, L2-resident), cvt A->bf16, double-buffered
// async-to-LDS GEMM (BK=64, 16 wmma/stage), online row softmax.
// ---------------------------------------------------------------------------

#define TOKENS 2048
#define IN_DIM 2048
#define VOCAB  32000

#define BM 128          // token rows per block
#define BN 128          // vocab cols per block
#define BK 64           // k per stage (two bf16 WMMA k-steps)
#define LDS_PITCH 72    // halfwords; 144B pitch -> conflict-free b128 reads
#define NSTAGE (IN_DIM / BK)   // 32

typedef __attribute__((ext_vector_type(16))) __bf16 v16bf;
typedef __attribute__((ext_vector_type(8)))  __bf16 bf16x8;
typedef __attribute__((ext_vector_type(8)))  float  v8f;
typedef __attribute__((ext_vector_type(4)))  int    v4i;

union BFrag { v16bf v; bf16x8 h[2]; };

// ---------------------------------------------------------------------------
// gfx1250 async global->LDS copy path (ASYNCcnt) with sync fallback
// Builtin signature (from hipcc diagnostic): param0 is AS(1) v4i*, so:
//   (global v4i* src, lds v4i* dst, imm offset, imm cpol)
// ---------------------------------------------------------------------------
#if __has_builtin(__builtin_amdgcn_global_load_async_to_lds_b128)
#define HAS_ASYNC_LDS 1
typedef __attribute__((address_space(1))) v4i* glb_v4i_p;
typedef __attribute__((address_space(3))) v4i* lds_v4i_p;
#else
#define HAS_ASYNC_LDS 0
#endif

__device__ __forceinline__ void copy_b128_to_lds(const unsigned short* g,
                                                 unsigned short* l) {
#if HAS_ASYNC_LDS
    __builtin_amdgcn_global_load_async_to_lds_b128(
        (glb_v4i_p)g, (lds_v4i_p)l, /*offset=*/0, /*cpol=*/0);
#else
    *(uint4*)l = *(const uint4*)g;   // global_load_b128 + ds_store_b128
#endif
}

__device__ __forceinline__ void wait_async_le8() {
#if HAS_ASYNC_LDS
#if __has_builtin(__builtin_amdgcn_s_wait_asynccnt)
    __builtin_amdgcn_s_wait_asynccnt(8);
#else
    asm volatile("s_wait_asynccnt 0x8" ::: "memory");
#endif
#endif
}

__device__ __forceinline__ void wait_async_le0() {
#if HAS_ASYNC_LDS
#if __has_builtin(__builtin_amdgcn_s_wait_asynccnt)
    __builtin_amdgcn_s_wait_asynccnt(0);
#else
    asm volatile("s_wait_asynccnt 0x0" ::: "memory");
#endif
#endif
}

// fp32 -> bf16 round-to-nearest-even, pure integer (no __bf16 scalar math)
__device__ __forceinline__ unsigned short f2bf(float f) {
    unsigned int u = __float_as_uint(f);
    u += 0x7FFFu + ((u >> 16) & 1u);
    return (unsigned short)(u >> 16);
}

// ---------------------------------------------------------------------------
// Kernel 1: fp32 -> bf16 staging (W: 131 MB bf16 -> fits 192 MB L2)
// ---------------------------------------------------------------------------
__global__ __launch_bounds__(256) void cvt_f32_to_bf16(
    const float4* __restrict__ in, ushort4* __restrict__ out, int n4)
{
    int i = blockIdx.x * blockDim.x + threadIdx.x;
    if (i >= n4) return;
    float4 v = in[i];
    ushort4 o;
    o.x = f2bf(v.x); o.y = f2bf(v.y); o.z = f2bf(v.z); o.w = f2bf(v.w);
    out[i] = o;
}

// ---------------------------------------------------------------------------
// Kernel 2: bf16 WMMA GEMM, C[row,col] = sum_k A[row,k] * W[col,k]
//   8 waves: 4 (M) x 2 (N); each wave: 32x64 = 2x4 tiles of v_wmma 16x16x32
//   Double-buffered LDS stages fed by async global->LDS copies.
// ---------------------------------------------------------------------------
__global__ __launch_bounds__(256) void lmhead_gemm_bf16(
    const unsigned short* __restrict__ A,   // [TOKENS][IN_DIM] bf16
    const unsigned short* __restrict__ W,   // [VOCAB][IN_DIM]  bf16
    float* __restrict__ out)                // [TOKENS][VOCAB]  logits
{
    __shared__ unsigned short sA[2][BM * LDS_PITCH];
    __shared__ unsigned short sW[2][BN * LDS_PITCH];

    const int tid  = threadIdx.x;
    const int wave = tid >> 5;
    const int lane = tid & 31;
    const int half = lane >> 4;   // which 16-lane half
    const int idx  = lane & 15;   // M (A-frag) / N (B-frag) index

    const int waveM = wave >> 1;  // 0..3
    const int waveN = wave & 1;   // 0..1

    const int rowBase = blockIdx.y * BM;
    const int colBase = blockIdx.x * BN;

    // cooperative tile-load assignment: 2 threads per 128B row, 64B each
    const int lr = tid >> 1;              // 0..127 tile row
    const int lc = (tid & 1) * 32;        // 0 or 32 (halfword column chunk)

    const unsigned short* gA = A + (size_t)(rowBase + lr) * IN_DIM + lc;
    const unsigned short* gW = W + (size_t)(colBase + lr) * IN_DIM + lc;
    unsigned short* dA0 = &sA[0][lr * LDS_PITCH + lc];
    unsigned short* dA1 = &sA[1][lr * LDS_PITCH + lc];
    unsigned short* dW0 = &sW[0][lr * LDS_PITCH + lc];
    unsigned short* dW1 = &sW[1][lr * LDS_PITCH + lc];

    const v8f vzero = {0.f, 0.f, 0.f, 0.f, 0.f, 0.f, 0.f, 0.f};
    v8f acc[2][4];
#pragma unroll
    for (int f = 0; f < 2; ++f)
#pragma unroll
        for (int g = 0; g < 4; ++g) acc[f][g] = vzero;

    // prologue: stage 0 into buffer 0
#pragma unroll
    for (int c = 0; c < 4; ++c) copy_b128_to_lds(gA + c * 8, dA0 + c * 8);
#pragma unroll
    for (int c = 0; c < 4; ++c) copy_b128_to_lds(gW + c * 8, dW0 + c * 8);

    for (int s = 0; s < NSTAGE; ++s) {
        const int buf = s & 1;
        if (s + 1 < NSTAGE) {
            // issue next stage into the other buffer (overlaps with compute)
            const int k0n = (s + 1) * BK;
            unsigned short* nA = (buf == 0) ? dA1 : dA0;
            unsigned short* nW = (buf == 0) ? dW1 : dW0;
#pragma unroll
            for (int c = 0; c < 4; ++c)
                copy_b128_to_lds(gA + k0n + c * 8, nA + c * 8);
#pragma unroll
            for (int c = 0; c < 4; ++c)
                copy_b128_to_lds(gW + k0n + c * 8, nW + c * 8);
            wait_async_le8();   // stage s's 8 copies retired (in-order)
        } else {
            wait_async_le0();
        }
        __syncthreads();        // stage s visible to all waves

        // two k-steps of 32 within the 64-wide stage
#pragma unroll
        for (int ks = 0; ks < 2; ++ks) {
            const int kc = ks * 32;
            // 16-bit A layout: lanes 0-15 K={0..7,16..23}; 16-31 K={8..15,24..31}
            BFrag a[2], b[4];
#pragma unroll
            for (int f = 0; f < 2; ++f) {
                const unsigned short* p =
                    &sA[buf][(waveM * 32 + f * 16 + idx) * LDS_PITCH + kc];
                a[f].h[0] = *(const bf16x8*)(p + 8 * half);
                a[f].h[1] = *(const bf16x8*)(p + 16 + 8 * half);
            }
#pragma unroll
            for (int g = 0; g < 4; ++g) {
                const unsigned short* p =
                    &sW[buf][(waveN * 64 + g * 16 + idx) * LDS_PITCH + kc];
                b[g].h[0] = *(const bf16x8*)(p + 8 * half);
                b[g].h[1] = *(const bf16x8*)(p + 16 + 8 * half);
            }
#pragma unroll
            for (int f = 0; f < 2; ++f)
#pragma unroll
                for (int g = 0; g < 4; ++g)
                    acc[f][g] = __builtin_amdgcn_wmma_f32_16x16x32_bf16(
                        /*neg_a=*/false, a[f].v, /*neg_b=*/false, b[g].v,
                        /*c_mod=*/(short)0, acc[f][g],
                        /*reuse_a=*/false, /*reuse_b=*/false);
        }

        if (s + 1 < NSTAGE)
            __syncthreads();    // all waves done reading buf before reuse
    }

    // writeback: C/D layout -> VGPR r: lanes0-15 M=r, lanes16-31 M=r+8; N=idx
#pragma unroll
    for (int f = 0; f < 2; ++f) {
#pragma unroll
        for (int g = 0; g < 4; ++g) {
            const int col = colBase + waveN * 64 + g * 16 + idx;
#pragma unroll
            for (int r = 0; r < 8; ++r) {
                const int row = rowBase + waveM * 32 + f * 16 + 8 * half + r;
                out[(size_t)row * VOCAB + col] = acc[f][g][r];
            }
        }
    }
}

// ---------------------------------------------------------------------------
// Kernel 3: in-place row softmax over VOCAB, online (m,s), 1 block per row
// ---------------------------------------------------------------------------
__global__ __launch_bounds__(256) void row_softmax(float* __restrict__ logits)
{
    const int row = blockIdx.x;
    float4* p4 = (float4*)(logits + (size_t)row * VOCAB);
    const int n4 = VOCAB / 4;   // 8000

    float m = -INFINITY, s = 0.f;
    for (int i = threadIdx.x; i < n4; i += 256) {
        float4 v = p4[i];
        float mx = fmaxf(fmaxf(v.x, v.y), fmaxf(v.z, v.w));
        if (mx > m) { s *= __expf(m - mx); m = mx; }
        s += __expf(v.x - m) + __expf(v.y - m)
           + __expf(v.z - m) + __expf(v.w - m);
    }
    // intra-wave (wave32) merge
#pragma unroll
    for (int off = 16; off > 0; off >>= 1) {
        float mo = __shfl_xor(m, off, 32);
        float so = __shfl_xor(s, off, 32);
        float M  = fmaxf(m, mo);
        s = s * __expf(m - M) + so * __expf(mo - M);
        m = M;
    }
    __shared__ float sm[8], ss[8];
    const int wv = threadIdx.x >> 5, ln = threadIdx.x & 31;
    if (ln == 0) { sm[wv] = m; ss[wv] = s; }
    __syncthreads();
    if (threadIdx.x == 0) {
        float M = sm[0];
        for (int i = 1; i < 8; ++i) M = fmaxf(M, sm[i]);
        float S = 0.f;
        for (int i = 0; i < 8; ++i) S += ss[i] * __expf(sm[i] - M);
        sm[0] = M;
        ss[0] = 1.f / S;
    }
    __syncthreads();
    const float M = sm[0], invS = ss[0];

    for (int i = threadIdx.x; i < n4; i += 256) {
        float4 v = p4[i];
        v.x = __expf(v.x - M) * invS;
        v.y = __expf(v.y - M) * invS;
        v.z = __expf(v.z - M) * invS;
        v.w = __expf(v.w - M) * invS;
        p4[i] = v;
    }
}

// ---------------------------------------------------------------------------
extern "C" void kernel_launch(void* const* d_in, const int* in_sizes, int n_in,
                              void* d_out, int out_size, void* d_ws, size_t ws_size,
                              hipStream_t stream) {
    const float* A32 = (const float*)d_in[0];   // input  [TOKENS, IN_DIM] fp32
    // d_in[1] = target (unused by reference forward)
    const float* W32 = (const float*)d_in[2];   // proj_weight [VOCAB, IN_DIM] fp32
    float* out = (float*)d_out;                 // [TOKENS, VOCAB] fp32

    // workspace: W_bf16 (131,072,000 B) then A_bf16 (8,388,608 B)
    unsigned short* Wb = (unsigned short*)d_ws;
    unsigned short* Ab = (unsigned short*)((char*)d_ws
                         + (size_t)VOCAB * IN_DIM * sizeof(unsigned short));

    {   // stage W -> bf16 (L2-resident across all 16 token-tiles)
        int n4 = VOCAB * IN_DIM / 4;
        cvt_f32_to_bf16<<<(n4 + 255) / 256, 256, 0, stream>>>(
            (const float4*)W32, (ushort4*)Wb, n4);
    }
    {   // stage A -> bf16
        int n4 = TOKENS * IN_DIM / 4;
        cvt_f32_to_bf16<<<(n4 + 255) / 256, 256, 0, stream>>>(
            (const float4*)A32, (ushort4*)Ab, n4);
    }

    dim3 gGemm(VOCAB / BN, TOKENS / BM);   // 250 x 16
    lmhead_gemm_bf16<<<gGemm, 256, 0, stream>>>(Ab, Wb, out);

    row_softmax<<<TOKENS, 256, 0, stream>>>(out);
}
